// MultiLoraConv2d_31525059953073
// MI455X (gfx1250) — matrix operand order repaired and checked by reading, not compiled, not run on hardware
//
#include <hip/hip_runtime.h>

// MultiLoraConv2d for MI455X (gfx1250), wave32.
//
//   K1 lora_synth : stack[t] = 2.0 * lora_B[t] @ lora_A[t]      (WMMA f32 16x16x4, exact)
//   K2 weight_agg : per-sample tap-major weights, split into bf16 hi/lo planes
//   K3 lora_conv  : per-sample conv as implicit GEMM, bf16 hi/lo split
//                   (3x v_wmma_f32_16x16x32_bf16 per K=32: hi*hi + hi*lo + lo*hi)
//                   A-tile staged with global_load_async_to_lds_b128 (ASYNCcnt)

#define NB   32     // batch
#define NCI  256
#define NCO  256
#define NH   56
#define NW   56
#define NT   8      // tasks
#define CK   768    // CIN*3
#define RK   48     // R*3
#define STACK_ELEMS (CK * CK)            // 589824 per task
#define AGG_ELEMS   ((size_t)NB * 9 * NCO * NCI)   // 18,874,368
#define SCALING 2.0f

typedef float  v2f  __attribute__((ext_vector_type(2)));
typedef float  v8f  __attribute__((ext_vector_type(8)));
typedef __bf16 v8bf __attribute__((ext_vector_type(8)));
typedef __bf16 v16bf __attribute__((ext_vector_type(16)));

// ---- bf16 split helpers (RNE, pure bit ops; data is finite) ----------------
static __device__ __forceinline__ unsigned short f32_to_bf16_rne(float f) {
    unsigned int u = __float_as_uint(f);
    u = (u + 0x7fffu + ((u >> 16) & 1u)) >> 16;
    return (unsigned short)u;
}
static __device__ __forceinline__ float bf16_bits_to_f32(unsigned short h) {
    return __uint_as_float(((unsigned int)h) << 16);
}

// ---- async global->LDS copy (CDNA5, tracked with ASYNCcnt) -----------------
// VDST VGPR = LDS byte offset; VADDR VGPR = 32-bit byte offset; SADDR = base.
static __device__ __forceinline__ void async_ld16(unsigned lds_byte_off,
                                                  const void* gbase,
                                                  unsigned gbyte_off) {
    asm volatile("global_load_async_to_lds_b128 %0, %1, %2"
                 :
                 : "v"(lds_byte_off), "v"(gbyte_off), "s"(gbase)
                 : "memory");
}
static __device__ __forceinline__ void wait_async0() {
    asm volatile("s_wait_asynccnt 0x0" ::: "memory");
}

// ---------------------------------------------------------------------------
// K1: stack[t] = 2 * lora_B[t](768x48) @ lora_A[t](48x768)  — exact fp32 WMMA
// ---------------------------------------------------------------------------
__global__ __launch_bounds__(256) void lora_synth(const float* __restrict__ loraB,
                                                  const float* __restrict__ loraA,
                                                  float* __restrict__ stack) {
    const int lane = threadIdx.x & 31;
    const int wid  = threadIdx.x >> 5;
    const int tile = blockIdx.x * 8 + wid;        // 0 .. 18431
    const int t  = tile / (48 * 48);
    const int rm = tile % (48 * 48);
    const int mt = rm / 48, nt = rm % 48;
    const int hi = lane >> 4;
    const int lm = lane & 15;

    const float* A  = loraB + (size_t)t * (CK * RK) + (size_t)(mt * 16 + lm) * RK;
    const float* Bm = loraA + (size_t)t * (RK * CK);
    const int col = nt * 16 + lm;

    v8f acc = {};
#pragma unroll
    for (int k = 0; k < RK; k += 4) {
        v2f af = *(const v2f*)(A + k + 2 * hi);
        v2f bf;
        bf.x = Bm[(k + hi) * CK + col];
        bf.y = Bm[(k + 2 + hi) * CK + col];
        acc = __builtin_amdgcn_wmma_f32_16x16x4_f32(false, af, false, bf,
                                                    (short)0, acc, false, false);
    }
    float* out = stack + (size_t)t * STACK_ELEMS;
#pragma unroll
    for (int r = 0; r < 8; ++r)
        out[(size_t)(mt * 16 + r + 8 * hi) * CK + col] = acc[r] * SCALING;
}

// ---------------------------------------------------------------------------
// K2: v = conv_w[o,ci,tap] + sum_t alpha[b,t]*stack[t][f]; store bf16 hi/lo
// Tap-major: aggT[b][tap][o][ci] -> contiguous 256x256 GEMM A-matrix per tap.
// ---------------------------------------------------------------------------
__global__ __launch_bounds__(256) void weight_agg(const float* __restrict__ convw,
                                                  const float* __restrict__ alphas,
                                                  const float* __restrict__ stack,
                                                  unsigned short* __restrict__ aggHi,
                                                  unsigned short* __restrict__ aggLo) {
    const int b   = blockIdx.y;
    const int tid = blockIdx.x * 256 + threadIdx.x;   // 0 .. 589823
    const int ci  = tid & 255;
    const int o   = (tid >> 8) & 255;
    const int tap = tid >> 16;                        // 0 .. 8
    const int f   = (o * 256 + ci) * 9 + tap;

    float v = convw[f];
    const float* al = alphas + b * NT;
#pragma unroll
    for (int t = 0; t < NT; ++t)
        v += al[t] * stack[(size_t)t * STACK_ELEMS + f];

    const size_t idx = (((size_t)b * 9 + tap) * 256 + o) * 256 + ci;
    unsigned short h = f32_to_bf16_rne(v);
    unsigned short l = f32_to_bf16_rne(v - bf16_bits_to_f32(h));
    aggHi[idx] = h;
    aggLo[idx] = l;
}

// ---------------------------------------------------------------------------
// K3: implicit-GEMM conv, bf16 split. Block = (Cout half m0, row h, sample b).
// M-tile 128 x N-tile 64; 8 waves as 4(M) x 2(N), each wave owns 32x32.
// ---------------------------------------------------------------------------
#define AP2 40   // LDS A row stride (bf16): 80 B rows -> b128-aligned, 2-way banks
#define BP2 40   // LDS B row stride (bf16, stored [n][k])

static __device__ __forceinline__ v16bf cat16(v8bf a, v8bf b) {
    return __builtin_shufflevector(a, b, 0, 1, 2, 3, 4, 5, 6, 7,
                                         8, 9, 10, 11, 12, 13, 14, 15);
}

__global__ __launch_bounds__(256) void lora_conv(const float* __restrict__ x,
                                                 const unsigned short* __restrict__ aggHi,
                                                 const unsigned short* __restrict__ aggLo,
                                                 float* __restrict__ out) {
    __shared__ __align__(16) unsigned short Ahs[128 * AP2];
    __shared__ __align__(16) unsigned short Als[128 * AP2];
    __shared__ __align__(16) unsigned short Bhs[64 * BP2];
    __shared__ __align__(16) unsigned short Bls[64 * BP2];

    const int b   = blockIdx.z;
    const int h   = blockIdx.y;
    const int m0  = blockIdx.x * 128;
    const int tid = threadIdx.x;
    const int lane = tid & 31, wid = tid >> 5;
    const int wm = wid & 3, wn = wid >> 2;       // wave position in 4x2 grid
    const int hi = lane >> 4, lm = lane & 15;

    // LDS byte offsets of the staging tiles (flat addr low 32 bits = LDS offset)
    const unsigned AhsB = (unsigned)(uintptr_t)(void*)Ahs;
    const unsigned AlsB = (unsigned)(uintptr_t)(void*)Als;

    v8f acc[2][2] = {{{}, {}}, {{}, {}}};
    const float* xb = x + (size_t)b * (NCI * NH * NW);

    for (int tap = 0; tap < 9; ++tap) {
        const int k1 = tap / 3, k2 = tap % 3;
        const int hin = h + k1 - 1;
        if (hin < 0 || hin >= NH) continue;      // uniform per block
        const size_t tapBase = ((size_t)b * 9 + tap) << 16;
        const unsigned short* Ah = aggHi + tapBase;
        const unsigned short* Al = aggLo + tapBase;

        for (int ci0 = 0; ci0 < NCI; ci0 += 32) {
            // ---- stage A asynchronously: 128(o) x 32(ci) bf16, hi & lo -----
#pragma unroll
            for (int i = 0; i < 2; ++i) {
                int i128 = i * 256 + tid;                // 16B chunk id, 0..511
                int orow = i128 >> 2;                    // 4 chunks per row
                int kc   = (i128 & 3) * 8;               // bf16 col
                unsigned g2 = (unsigned)(((m0 + orow) * 256 + ci0 + kc) * 2);
                unsigned l2 = (unsigned)((orow * AP2 + kc) * 2);
                async_ld16(AhsB + l2, Ah, g2);
                async_ld16(AlsB + l2, Al, g2);
            }
            // ---- stage B: x row -> bf16 hi/lo, transposed [n][ci] ----------
#pragma unroll
            for (int i = 0; i < 8; ++i) {
                int idx = i * 256 + tid;                 // 0..2047
                int kc  = idx >> 6;
                int n   = idx & 63;
                int win = n + k2 - 1;
                float v = 0.0f;
                if (win >= 0 && win < NW)
                    v = xb[((size_t)(ci0 + kc) * NH + hin) * NW + win];
                unsigned short hh = f32_to_bf16_rne(v);
                unsigned short ll = f32_to_bf16_rne(v - bf16_bits_to_f32(hh));
                Bhs[n * BP2 + kc] = hh;
                Bls[n * BP2 + kc] = ll;
            }
            wait_async0();                               // our async copies landed
            __syncthreads();                             // everyone's landed

            // ---- fragments (ISA 16-bit layouts) ----------------------------
            // A 16x32: lane (hi,lm): K = 8*hi+{0..7} and 8*hi+16+{0..7}
            // B 32x16 (stored [n][k]): lane: K = 8*hi+{0..7} and 8*hi+16+{0..7}
            v16bf ah[2], al2[2], bh[2], bl[2];
#pragma unroll
            for (int ms = 0; ms < 2; ++ms) {
                int row = (wm * 32 + ms * 16 + lm) * AP2 + 8 * hi;
                ah[ms]  = cat16(*(const v8bf*)(&Ahs[row]),
                                *(const v8bf*)(&Ahs[row + 16]));
                al2[ms] = cat16(*(const v8bf*)(&Als[row]),
                                *(const v8bf*)(&Als[row + 16]));
            }
#pragma unroll
            for (int ns = 0; ns < 2; ++ns) {
                int rowb = (wn * 32 + ns * 16 + lm) * BP2 + 8 * hi;
                bh[ns] = cat16(*(const v8bf*)(&Bhs[rowb]),
                               *(const v8bf*)(&Bhs[rowb + 16]));
                bl[ns] = cat16(*(const v8bf*)(&Bls[rowb]),
                               *(const v8bf*)(&Bls[rowb + 16]));
            }
            // ---- 12 x v_wmma_f32_16x16x32_bf16 (hi*hi + hi*lo + lo*hi) -----
#pragma unroll
            for (int ms = 0; ms < 2; ++ms)
#pragma unroll
                for (int ns = 0; ns < 2; ++ns) {
                    v8f c = acc[ms][ns];
                    c = __builtin_amdgcn_wmma_f32_16x16x32_bf16(false, ah[ms],  false, bh[ns],
                                                                (short)0, c, false, false);
                    c = __builtin_amdgcn_wmma_f32_16x16x32_bf16(false, ah[ms],  false, bl[ns],
                                                                (short)0, c, false, false);
                    c = __builtin_amdgcn_wmma_f32_16x16x32_bf16(false, al2[ms], false, bh[ns],
                                                                (short)0, c, false, false);
                    acc[ms][ns] = c;
                }
            __syncthreads();
        }
    }

    // Epilogue: C layout — lane (hi,lm), VGPR r -> (M = r + 8*hi, N = lm)
    float* ob = out + ((size_t)b * NCO + m0) * (NH * NW) + h * NW;
#pragma unroll
    for (int ms = 0; ms < 2; ++ms) {
#pragma unroll
        for (int ns = 0; ns < 2; ++ns) {
            int w = wn * 32 + ns * 16 + lm;
            if (w < NW) {
#pragma unroll
                for (int r = 0; r < 8; ++r) {
                    int m = wm * 32 + ms * 16 + r + 8 * hi;
                    ob[(size_t)m * (NH * NW) + w] = acc[ms][ns][r];
                }
            }
        }
    }
}

// ---------------------------------------------------------------------------
extern "C" void kernel_launch(void* const* d_in, const int* in_sizes, int n_in,
                              void* d_out, int out_size, void* d_ws, size_t ws_size,
                              hipStream_t stream) {
    const float* x      = (const float*)d_in[0];   // [32,256,56,56]
    const float* alphas = (const float*)d_in[1];   // [32,8]
    const float* convw  = (const float*)d_in[2];   // [256,256,3,3]
    const float* loraA  = (const float*)d_in[3];   // [8,48,768]
    const float* loraB  = (const float*)d_in[4];   // [8,768,48]
    float* out = (float*)d_out;                    // [32,256,56,56]

    float* stack = (float*)d_ws;                               // 18.9 MB fp32
    unsigned short* aggHi = (unsigned short*)(stack + (size_t)NT * STACK_ELEMS); // 37.7 MB
    unsigned short* aggLo = aggHi + AGG_ELEMS;                                   // 37.7 MB

    lora_synth<<<dim3(2304), dim3(256), 0, stream>>>(loraB, loraA, stack);
    weight_agg<<<dim3(2304, 32), dim3(256), 0, stream>>>(convw, alphas, stack, aggHi, aggLo);
    lora_conv<<<dim3(2, 56, 32), dim3(256), 0, stream>>>(x, aggHi, aggLo, out);
}
